// ClassicAttention_24300924961352
// MI455X (gfx1250) — matrix operand order repaired and checked by reading, not compiled, not run on hardware
//
#include <hip/hip_runtime.h>
#include <hip/hip_bf16.h>
#include <cstdint>

typedef __bf16 bf16_t;
typedef __attribute__((ext_vector_type(8)))  __bf16 v8bf;
typedef __attribute__((ext_vector_type(16))) __bf16 v16bf;
typedef __attribute__((ext_vector_type(8)))  float  v8f;

#define TK  32
#define PAD 8   // 16B: keeps LDS rows 16B-aligned, staggers banks

// ---------------------------------------------------------------- WMMA helpers
__device__ __forceinline__ v16bf frag_cat(const bf16_t* p0, const bf16_t* p1) {
  v8bf lo = *(const v8bf*)p0;
  v8bf hi = *(const v8bf*)p1;
  return __builtin_shufflevector(lo, hi, 0,1,2,3,4,5,6,7,8,9,10,11,12,13,14,15);
}
// A-operand (16x32, row-major). rp -> &src[row][kb]. Halves hold K {0-7,16-23}/{8-15,24-31}
__device__ __forceinline__ v16bf frag_a(const bf16_t* rp, int lane) {
  const int kh = (lane < 16) ? 0 : 8;
  return frag_cat(rp + kh, rp + kh + 16);
}
// B-operand (32x16, src N-major: row n holds a K-run). Halves hold K {0-7,16-23}/{8-15,24-31}
__device__ __forceinline__ v16bf frag_b(const bf16_t* rp, int lane) {
  const int ko = (lane < 16) ? 0 : 8;
  return frag_cat(rp + ko, rp + ko + 16);
}
__device__ __forceinline__ v8f wmma_bf16(v16bf a, v16bf b, v8f c) {
  return __builtin_amdgcn_wmma_f32_16x16x32_bf16(false, a, false, b, (short)0, c,
                                                 false, false);
}

// ---------------------------------------------------------------- TDM (async tensor DMA)
#if __has_builtin(__builtin_amdgcn_tensor_load_to_lds)
#define HAVE_TDM 1
#else
#define HAVE_TDM 0
#endif

#if HAVE_TDM
typedef __attribute__((ext_vector_type(4))) unsigned int v4u;
typedef __attribute__((ext_vector_type(4))) int          v4i;
typedef __attribute__((ext_vector_type(8))) int          v8i;

// 2D bf16 tile load: global (row stride = stride0 elems) -> LDS, with optional
// per-row padding so LDS rows can carry a bank-stagger pad.
__device__ __forceinline__ void tdm_load_2d_bf16(
    const void* gaddr, unsigned lds_byte_addr,
    unsigned tile0, unsigned tile1,          // elems per row, rows
    unsigned tensor0, unsigned tensor1,      // tensor extents (OOB clamp)
    unsigned stride0,                        // elems between rows
    unsigned pad_interval_code, unsigned pad_amount_code, unsigned pad_en) {
  const unsigned long long ga = (unsigned long long)gaddr;
  v4u g0;
  g0[0] = 1u;                                        // count=1, user mode
  g0[1] = lds_byte_addr;                             // lds_addr
  g0[2] = (unsigned)(ga & 0xFFFFFFFFu);              // global_addr[31:0]
  g0[3] = (unsigned)((ga >> 32) & 0x01FFFFFFu) | (2u << 30);  // addr[56:32] | type=2
  v8i g1;
  g1[0] = (int)((1u << 16) |                         // data_size = 2B
                (pad_en ? ((1u << 20) | (pad_interval_code << 22) |
                           (pad_amount_code << 25)) : 0u));
  g1[1] = (int)((tensor0 & 0xFFFFu) << 16);          // tensor_dim0[15:0]
  g1[2] = (int)(((tensor0 >> 16) & 0xFFFFu) | ((tensor1 & 0xFFFFu) << 16));
  g1[3] = (int)(((tensor1 >> 16) & 0xFFFFu) | ((tile0 & 0xFFFFu) << 16));
  g1[4] = (int)(tile1 & 0xFFFFu);                    // tile_dim1 (tile_dim2=0)
  g1[5] = (int)stride0;                              // tensor_dim0_stride[31:0]
  g1[6] = 0;                                         // stride hi / dim1_stride lo
  g1[7] = 0;
  v4i z4 = {0, 0, 0, 0};
#if __clang_major__ >= 23
  v8i z8 = {0, 0, 0, 0, 0, 0, 0, 0};
  __builtin_amdgcn_tensor_load_to_lds(g0, g1, z4, z4, z8, 0);
#else
  __builtin_amdgcn_tensor_load_to_lds(g0, g1, z4, z4, 0);
#endif
}
#endif  // HAVE_TDM

// Stage a 128x32 bf16 X tile into an LDS buffer with PAD-element row padding.
__device__ __forceinline__ void stage_x(const bf16_t* __restrict__ Xtile, int ldx,
                                        int rowsTotal, bf16_t (*buf)[TK + PAD],
                                        int t, int wave) {
#if HAVE_TDM
  if (wave == 0) {
    // pad_interval: 16 DW (=64B row) -> code 3 ; pad_amount: 4 DW (=16B) -> code 3
    tdm_load_2d_bf16(Xtile, (unsigned)(uintptr_t)&buf[0][0],
                     TK, 128, (unsigned)ldx, (unsigned)rowsTotal, (unsigned)ldx,
                     3u, 3u, 1u);
  }
#else
  (void)rowsTotal;
#pragma unroll
  for (int i = 0; i < 2; ++i) {
    const int c = t + i * 256;
    const int r = c >> 2, col = (c & 3) * 8;
    *(v8bf*)&buf[r][col] = *(const v8bf*)(Xtile + (size_t)r * ldx + col);
  }
#endif
}
__device__ __forceinline__ void stage_wait(int wave) {
#if HAVE_TDM
  if (wave == 0) __builtin_amdgcn_s_wait_tensorcnt(0);
#endif
  (void)wave;
}

// ---------------------------------------------------------------- conversion kernels
__global__ void f32_to_bf16_kernel(const float* __restrict__ src,
                                   bf16_t* __restrict__ dst, int n) {
  int i = blockIdx.x * blockDim.x + threadIdx.x;
  if (i < n) dst[i] = (bf16_t)src[i];
}

// dst[n][k] = (bf16) src[k][n]   (K x N fp32 -> N x K bf16)
__global__ __launch_bounds__(256) void transpose_f32_to_bf16_kernel(
    const float* __restrict__ src, bf16_t* __restrict__ dst, int K, int N) {
  __shared__ float tile[32][33];
  const int n0 = blockIdx.x * 32, k0 = blockIdx.y * 32;
  const int tx = threadIdx.x & 31, ty = threadIdx.x >> 5;
#pragma unroll
  for (int i = 0; i < 32; i += 8)
    tile[ty + i][tx] = src[(size_t)(k0 + ty + i) * N + n0 + tx];
  __syncthreads();
#pragma unroll
  for (int i = 0; i < 32; i += 8)
    dst[(size_t)(n0 + ty + i) * K + k0 + tx] = (bf16_t)tile[tx][ty + i];
}

// ---------------------------------------------------------------- GEMM
// C[M,N] = X[M,K] @ Wt[N,K]^T + bias.  X tile double-buffered in LDS via TDM;
// B-fragments stream from global (Wt is L2-resident).
// MODE 0: fp32 out.  MODE 1: scatter bf16 into per-head Q/K/V [B,H,S,64].
template <int MODE>
__global__ __launch_bounds__(256) void gemm_bf16_kernel(
    const bf16_t* __restrict__ X, const bf16_t* __restrict__ Wt,
    const float* __restrict__ bias, int M, int N, int Kdim,
    float* __restrict__ outF,
    bf16_t* __restrict__ Qb, bf16_t* __restrict__ Kb, bf16_t* __restrict__ Vb) {
  __shared__ bf16_t Xs[2][128][TK + PAD];
  const int t = threadIdx.x, lane = t & 31, wave = t >> 5;
  const int nBase = blockIdx.x * 128, mBase = blockIdx.y * 128;
  const int waveM = wave >> 2, waveN = wave & 3;  // 2x4 waves, wave tile 64x32
  v8f acc[4][2] = {};

  const bf16_t* Xblk = X + (size_t)mBase * Kdim;

  stage_x(Xblk, Kdim, M, Xs[0], t, wave);
  stage_wait(wave);
  __syncthreads();

  const int nSteps = Kdim / TK;
  for (int s = 0; s < nSteps; ++s) {
    const int kb = s * TK;
    const int cur = s & 1;
    if (s + 1 < nSteps) {
      stage_x(Xblk + kb + TK, Kdim, M, Xs[cur ^ 1], t, wave);  // async overlap
      __builtin_prefetch(Wt + (size_t)(nBase + (t >> 1)) * Kdim + kb + TK +
                             (t & 1) * 16, 0, 1);
    }

    v16bf af[4], bfr[2];
#pragma unroll
    for (int ni = 0; ni < 2; ++ni)
      bfr[ni] = frag_b(
          Wt + (size_t)(nBase + waveN * 32 + ni * 16 + (lane & 15)) * Kdim + kb,
          lane);
#pragma unroll
    for (int mi = 0; mi < 4; ++mi)
      af[mi] = frag_a(&Xs[cur][waveM * 64 + mi * 16 + (lane & 15)][0], lane);
#pragma unroll
    for (int mi = 0; mi < 4; ++mi)
#pragma unroll
      for (int ni = 0; ni < 2; ++ni)
        acc[mi][ni] = wmma_bf16(af[mi], bfr[ni], acc[mi][ni]);

    stage_wait(wave);
    __syncthreads();
  }

#pragma unroll
  for (int mi = 0; mi < 4; ++mi) {
#pragma unroll
    for (int ni = 0; ni < 2; ++ni) {
      const int n = nBase + waveN * 32 + ni * 16 + (lane & 15);
      const float bv = bias[n];
#pragma unroll
      for (int r = 0; r < 8; ++r) {
        const int m = mBase + waveM * 64 + mi * 16 + r + ((lane < 16) ? 0 : 8);
        const float val = acc[mi][ni][r] + bv;
        if (MODE == 0) {
          outF[(size_t)m * N + n] = val;
        } else {
          const int which = n >> 10, d = n & 1023, h = d >> 6, e = d & 63;
          const int b = m >> 11, s2 = m & 2047;
          bf16_t* dst = (which == 0) ? Qb : (which == 1) ? Kb : Vb;
          dst[((((size_t)b * 16 + h) * 2048 + s2) << 6) + e] = (bf16_t)val;
        }
      }
    }
  }
}

// ---------------------------------------------------------------- flash attention
// One block = (b,h, 128 query rows); 8 waves x 16-row strips.
__global__ __launch_bounds__(256) void attn_kernel(
    const bf16_t* __restrict__ Q, const bf16_t* __restrict__ K,
    const bf16_t* __restrict__ V, bf16_t* __restrict__ CTX) {
  constexpr int S = 2048, HD = 64;
  __shared__ bf16_t Vt[64][128 + PAD];     // V tile transposed [hd][key]
  __shared__ bf16_t Ps[8][16][128 + PAD];  // per-wave P strip
  const int t = threadIdx.x, lane = t & 31, wave = t >> 5;
  const int qTile = blockIdx.x * 128;
  const int bh = blockIdx.y;
  const bf16_t* Qh = Q + (size_t)bh * S * HD;
  const bf16_t* Kh = K + (size_t)bh * S * HD;
  const bf16_t* Vh = V + (size_t)bh * S * HD;
  const int qRow0 = qTile + wave * 16;
  const int rOff = (lane < 16) ? 0 : 8;

  float m_r[8], l_r[8];
  v8f o[4] = {};
#pragma unroll
  for (int r = 0; r < 8; ++r) { m_r[r] = -3.0e38f; l_r[r] = 0.f; }

  v16bf qf[2];  // loop-invariant Q fragments (hd=64 -> 2 k-steps)
#pragma unroll
  for (int ks = 0; ks < 2; ++ks)
    qf[ks] = frag_a(Qh + (size_t)(qRow0 + (lane & 15)) * HD + ks * 32, lane);

  const int kEnd = qTile + 128;  // causal limit
  for (int kb = 0; kb < kEnd; kb += 128) {
    __syncthreads();  // previous iteration done reading Vt
#pragma unroll
    for (int i = 0; i < 4; ++i) {  // stage V tile transposed
      const int c = t + i * 256;
      const int row = c >> 3, e0 = (c & 7) * 8;
      v8bf v = *(const v8bf*)(Vh + (size_t)(kb + row) * HD + e0);
#pragma unroll
      for (int j = 0; j < 8; ++j) Vt[e0 + j][row] = v[j];
    }
    if (kb + 128 < kEnd) {  // prefetch next K/V tiles into cache
      __builtin_prefetch(Kh + (size_t)(kb + 128 + (t >> 1)) * HD + (t & 1) * 32,
                         0, 1);
      __builtin_prefetch(Vh + (size_t)(kb + 128 + (t >> 1)) * HD + (t & 1) * 32,
                         0, 1);
    }
    __syncthreads();

    // scores strip 16x128: K rows feed B-fragments straight from global
    v8f s[8] = {};
#pragma unroll
    for (int j = 0; j < 8; ++j) {
      const bf16_t* krow = Kh + (size_t)(kb + j * 16 + (lane & 15)) * HD;
      v16bf kf0 = frag_b(krow, lane);
      v16bf kf1 = frag_b(krow + 32, lane);
      s[j] = wmma_bf16(qf[0], kf0, s[j]);
      s[j] = wmma_bf16(qf[1], kf1, s[j]);
    }

    // online softmax (row stats redundant across each 16-lane half)
    float alpha[8];
#pragma unroll
    for (int r = 0; r < 8; ++r) {
      const int qr = qRow0 + r + rOff;
      float mx = -3.0e38f;
#pragma unroll
      for (int j = 0; j < 8; ++j) {
        const int col = kb + j * 16 + (lane & 15);
        float sv = s[j][r] * 0.125f;        // 1/sqrt(64)
        sv = (col <= qr) ? sv : -3.0e38f;   // causal mask
        s[j][r] = sv;
        mx = fmaxf(mx, sv);
      }
#pragma unroll
      for (int d2 = 1; d2 < 16; d2 <<= 1) mx = fmaxf(mx, __shfl_xor(mx, d2));
      const float mnew = fmaxf(m_r[r], mx);
      alpha[r] = __expf(m_r[r] - mnew);
      float rs = 0.f;
#pragma unroll
      for (int j = 0; j < 8; ++j) {
        const float p = __expf(s[j][r] - mnew);
        s[j][r] = p;
        rs += p;
      }
#pragma unroll
      for (int d2 = 1; d2 < 16; d2 <<= 1) rs += __shfl_xor(rs, d2);
      l_r[r] = l_r[r] * alpha[r] + rs;
      m_r[r] = mnew;
    }
#pragma unroll
    for (int jo = 0; jo < 4; ++jo)
#pragma unroll
      for (int r = 0; r < 8; ++r) o[jo][r] *= alpha[r];

    // restage P through LDS (D-layout -> A-fragment layout)
#pragma unroll
    for (int r = 0; r < 8; ++r)
#pragma unroll
      for (int j = 0; j < 8; ++j)
        Ps[wave][r + rOff][j * 16 + (lane & 15)] = (bf16_t)s[j][r];

    // O += P @ V  (128 keys -> 4 k-steps)
#pragma unroll
    for (int kk = 0; kk < 4; ++kk) {
      v16bf pf = frag_a(&Ps[wave][lane & 15][kk * 32], lane);
#pragma unroll
      for (int jo = 0; jo < 4; ++jo) {
        v16bf vf = frag_b(&Vt[jo * 16 + (lane & 15)][kk * 32], lane);
        o[jo] = wmma_bf16(pf, vf, o[jo]);
      }
    }
  }

  // normalize, write ctx merged-head [B,S,H*64]
  const int b = bh >> 4, h = bh & 15;
#pragma unroll
  for (int jo = 0; jo < 4; ++jo) {
    const int e = jo * 16 + (lane & 15);
#pragma unroll
    for (int r = 0; r < 8; ++r) {
      const int srow = qRow0 + r + rOff;
      const float val = o[jo][r] / l_r[r];
      CTX[(((size_t)b * S + srow) * 16 + h) * 64 + e] = (bf16_t)val;
    }
  }
}

// ---------------------------------------------------------------- launcher
extern "C" void kernel_launch(void* const* d_in, const int* in_sizes, int n_in,
                              void* d_out, int out_size, void* d_ws, size_t ws_size,
                              hipStream_t stream) {
  (void)in_sizes; (void)n_in; (void)out_size; (void)ws_size;
  const float* hidden = (const float*)d_in[0];
  const float* attn_w = (const float*)d_in[1];
  const float* attn_b = (const float*)d_in[2];
  const float* proj_w = (const float*)d_in[3];
  const float* proj_b = (const float*)d_in[4];
  float* out = (float*)d_out;

  constexpr int B = 2, S = 2048, D = 1024, H = 16;
  constexpr int M = B * S;  // 4096

  size_t off = 0;
  char* ws = (char*)d_ws;
  auto alloc = [&](size_t bytes) -> char* {
    char* p = ws + off;
    off += (bytes + 255) & ~(size_t)255;
    return p;
  };
  bf16_t* Xbf   = (bf16_t*)alloc((size_t)M * D * 2);
  bf16_t* WqkvT = (bf16_t*)alloc((size_t)3 * D * D * 2);  // [3D][D]
  bf16_t* WpT   = (bf16_t*)alloc((size_t)D * D * 2);      // [D][D]
  bf16_t* Qb    = (bf16_t*)alloc((size_t)M * D * 2);
  bf16_t* Kb    = (bf16_t*)alloc((size_t)M * D * 2);
  bf16_t* Vb    = (bf16_t*)alloc((size_t)M * D * 2);
  bf16_t* CTX   = (bf16_t*)alloc((size_t)M * D * 2);

  f32_to_bf16_kernel<<<(M * D + 255) / 256, 256, 0, stream>>>(hidden, Xbf, M * D);
  transpose_f32_to_bf16_kernel<<<dim3(3 * D / 32, D / 32), 256, 0, stream>>>(
      attn_w, WqkvT, D, 3 * D);
  transpose_f32_to_bf16_kernel<<<dim3(D / 32, D / 32), 256, 0, stream>>>(
      proj_w, WpT, D, D);

  // QKV projection -> per-head Q/K/V (bf16)
  gemm_bf16_kernel<1><<<dim3(3 * D / 128, M / 128), 256, 0, stream>>>(
      Xbf, WqkvT, attn_b, M, 3 * D, D, nullptr, Qb, Kb, Vb);

  // flash attention: 32 (b,h) pairs x 16 q-tiles
  attn_kernel<<<dim3(S / 128, B * H), 256, 0, stream>>>(Qb, Kb, Vb, CTX);

  // output projection -> fp32
  gemm_bf16_kernel<0><<<dim3(D / 128, M / 128), 256, 0, stream>>>(
      CTX, WpT, proj_b, M, D, D, out, nullptr, nullptr, nullptr);
}
